// ImageSelfAttention_83202106458615
// MI455X (gfx1250) — matrix operand order repaired and checked by reading, not compiled, not run on hardware
//
#include <hip/hip_runtime.h>

typedef __attribute__((ext_vector_type(16))) _Float16 v16h;
typedef __attribute__((ext_vector_type(8)))  _Float16 v8h;
typedef __attribute__((ext_vector_type(8)))  float    v8f;

#define B_   32
#define C_   256
#define HW_  1024
#define CPG_ 8   // channels per group (32 groups)

// ---------------------------------------------------------------- WMMA helpers

__device__ __forceinline__ v8f wmma16(v16h a, v16h b, v8f c) {
  // D = A(16x32 f16) * B(32x16 f16) + C(16x16 f32)
  return __builtin_amdgcn_wmma_f32_16x16x32_f16(false, a, false, b, (short)0, c,
                                                false, false);
}

// A-fragment (16x32 f16) from row-major [m][k] memory (ld in halves).
// ISA layout: lanes 0-15 = M 0-15 holding K {0-7,16-23}; lanes 16-31 hold K {8-15,24-31}.
__device__ __forceinline__ v16h load_a(const _Float16* base, int ld) {
  const int lane = threadIdx.x & 31;
  const _Float16* p = base + (lane & 15) * ld + ((lane >> 4) << 3);
  v8h lo = *(const v8h*)p;          // K = kk .. kk+7
  v8h hi = *(const v8h*)(p + 16);   // K = kk+16 .. kk+23
  v16h r;
#pragma unroll
  for (int i = 0; i < 8; ++i) { r[i] = lo[i]; r[i + 8] = hi[i]; }
  return r;
}

// B-fragment (32x16 f16); element (k, col) lives at base[col*ld + k].
// ISA layout: lane%16 = column; lanes 0-15 hold K {0-7,16-23}, lanes 16-31 hold K {8-15,24-31}.
__device__ __forceinline__ v16h load_b(const _Float16* base, int ld) {
  const int lane = threadIdx.x & 31;
  const _Float16* p = base + (lane & 15) * ld + ((lane >> 4) << 3);
  v8h lo = *(const v8h*)p;
  v8h hi = *(const v8h*)(p + 16);
  v16h r;
#pragma unroll
  for (int i = 0; i < 8; ++i) { r[i] = lo[i]; r[i + 8] = hi[i]; }
  return r;
}

// ---------------------------------------------------------------- kernel 0: weights -> f16

__global__ void k_w2h(const float* __restrict__ wq, const float* __restrict__ wk,
                      const float* __restrict__ wv, const float* __restrict__ wo,
                      _Float16* __restrict__ w16) {
  const int i = blockIdx.x * 256 + threadIdx.x;       // grid 256 -> i < 65536
  w16[i]          = (_Float16)wq[i];
  w16[65536 + i]  = (_Float16)wk[i];
  w16[131072 + i] = (_Float16)wv[i];
  w16[196608 + i] = (_Float16)wo[i];
}

// ---------------------------------------------------------------- kernel 1: GroupNorm -> hn[B][N][C] f16

__global__ void k_groupnorm(const float* __restrict__ x,
                            const float* __restrict__ gs,
                            const float* __restrict__ gb,
                            _Float16* __restrict__ hn) {
  const int b = blockIdx.x >> 5;
  const int g = blockIdx.x & 31;
  const int t = threadIdx.x;
  const float* xb = x + (b * C_ + g * CPG_) * HW_;

  float s = 0.f, ss = 0.f;
#pragma unroll
  for (int i = t; i < CPG_ * HW_; i += 256) {
    float v = xb[i];
    s += v; ss += v * v;
  }
#pragma unroll
  for (int m = 16; m; m >>= 1) {
    s  += __shfl_xor(s,  m, 32);
    ss += __shfl_xor(ss, m, 32);
  }
  __shared__ float red[16];
  if ((t & 31) == 0) { red[t >> 5] = s; red[8 + (t >> 5)] = ss; }
  __syncthreads();
  s = 0.f; ss = 0.f;
#pragma unroll
  for (int w = 0; w < 8; ++w) { s += red[w]; ss += red[8 + w]; }

  const float mean = s * (1.f / 8192.f);
  const float var  = ss * (1.f / 8192.f) - mean * mean;
  const float rstd = rsqrtf(var + 1e-6f);

  float sc[CPG_], bi[CPG_];
#pragma unroll
  for (int cl = 0; cl < CPG_; ++cl) {
    sc[cl] = gs[g * CPG_ + cl] * rstd;
    bi[cl] = gb[g * CPG_ + cl] - mean * sc[cl];
  }
  // transpose store: 8 channels of this group packed per pixel -> 16B store
#pragma unroll
  for (int nb = 0; nb < 4; ++nb) {
    const int n = nb * 256 + t;
    v8h o;
#pragma unroll
    for (int cl = 0; cl < CPG_; ++cl)
      o[cl] = (_Float16)(xb[cl * HW_ + n] * sc[cl] + bi[cl]);
    *(v8h*)(hn + ((b * HW_ + n) << 8) + g * CPG_) = o;
  }
}

// ---------------------------------------------------------------- kernel 2: fused Q/K/V projections
// q,k stored [B][N][C] (C contiguous); v stored [B][C][N] (N contiguous).
// log2(e)/sqrt(C) folded into q so attention can use exp2 directly.

__global__ void __launch_bounds__(256)
__attribute__((amdgpu_waves_per_eu(2)))
k_qkv(const _Float16* __restrict__ w16,
      const _Float16* __restrict__ hn,
      const float* __restrict__ bq, const float* __restrict__ bk,
      const float* __restrict__ bv,
      _Float16* __restrict__ q, _Float16* __restrict__ k,
      _Float16* __restrict__ v) {
  const int b    = blockIdx.x >> 3;
  const int nt   = blockIdx.x & 7;
  const int w    = threadIdx.x >> 5;
  const int lane = threadIdx.x & 31;
  const int n0   = nt * 128 + w * 16;

  const _Float16* hb = hn + (b * HW_ + n0) * C_;
  v16h bf[8];                                   // hn B-frags, K=256, reused 48x
#pragma unroll
  for (int kt = 0; kt < 8; ++kt) bf[kt] = load_b(hb + kt * 32, C_);

  const float sq = 0.0901684400555602f;         // log2(e) / sqrt(256)
  const int nn = n0 + (lane & 15);
  const int mh = (lane >> 4) << 3;

  for (int ot = 0; ot < 16; ++ot) {
    const int o0 = ot * 16;
    v8f aq = {}, ak = {}, av = {};
#pragma unroll
    for (int kt = 0; kt < 8; ++kt) {
      v16h wa;
      wa = load_a(w16 +          o0 * C_ + kt * 32, C_); aq = wmma16(wa, bf[kt], aq);
      wa = load_a(w16 +  65536 + o0 * C_ + kt * 32, C_); ak = wmma16(wa, bf[kt], ak);
      wa = load_a(w16 + 131072 + o0 * C_ + kt * 32, C_); av = wmma16(wa, bf[kt], av);
    }
    v8h qv, kv;
#pragma unroll
    for (int r = 0; r < 8; ++r) {
      const int o = o0 + mh + r;
      qv[r] = (_Float16)((aq[r] + bq[o]) * sq);
      kv[r] = (_Float16)(ak[r] + bk[o]);
      v[(b * C_ + o) * HW_ + nn] = (_Float16)(av[r] + bv[o]);
    }
    // D-tile lane holds 8 consecutive output channels for one pixel -> 16B store
    *(v8h*)(q + (b * HW_ + nn) * C_ + o0 + mh) = qv;
    *(v8h*)(k + (b * HW_ + nn) * C_ + o0 + mh) = kv;
  }
}

// ---------------------------------------------------------------- kernel 3: flash attention
// 128-thread blocks (4 waves = 1 wave/SIMD) + waves_per_eu(1): full register
// file available so the 64-VGPR Q fragments and 128-VGPR f32 accumulator stay
// resident (no scratch spills in the j-loop). Per wave: 16 query rows.

__global__ void __launch_bounds__(128)
__attribute__((amdgpu_waves_per_eu(1)))
k_attn(const _Float16* __restrict__ q,
       const _Float16* __restrict__ kmat,
       const _Float16* __restrict__ vmat,
       _Float16* __restrict__ o2) {
  const int b    = blockIdx.x >> 4;
  const int it   = blockIdx.x & 15;
  const int w    = threadIdx.x >> 5;            // 0..3
  const int lane = threadIdx.x & 31;
  const int i0   = it * 64 + w * 16;

  __shared__ _Float16 P[4 * 16 * 32];           // per-wave 16x32 P-tile bounce
  _Float16* pw = P + w * 512;

  v16h qf[8];
  const _Float16* qb = q + (b * HW_ + i0) * C_;
#pragma unroll
  for (int ct = 0; ct < 8; ++ct) qf[ct] = load_a(qb + ct * 32, C_);

  const v8f vzero = {};
  v8f acc[16];                                  // 16 rows x 256 cols f32
#pragma unroll
  for (int ct = 0; ct < 16; ++ct) acc[ct] = vzero;
  float mrow[8], lrow[8];
#pragma unroll
  for (int r = 0; r < 8; ++r) { mrow[r] = -1e30f; lrow[r] = 0.f; }

  const _Float16* kb = kmat + b * HW_ * C_;
  const _Float16* vb = vmat + b * C_ * HW_;
  _Float16*       pst = pw + (((lane >> 4) << 3) * 32) + (lane & 15);  // C/D -> [i][j]
  const _Float16* pld = pw + (lane & 15) * 32 + ((lane >> 4) << 3);    // A-frag read

  for (int j0 = 0; j0 < HW_; j0 += 32) {
    // ---- scores: two 16x16 tiles over K=256 (already in log2 domain)
    v8f s0 = vzero, s1 = vzero;
#pragma unroll
    for (int ct = 0; ct < 8; ++ct) {
      s0 = wmma16(qf[ct], load_b(kb + j0 * C_ + ct * 32, C_), s0);
      s1 = wmma16(qf[ct], load_b(kb + (j0 + 16) * C_ + ct * 32, C_), s1);
    }
    if (j0 + 32 < HW_) {                        // global_prefetch_b8 next K tile
      __builtin_prefetch(kb + (j0 + 32) * C_, 0, 3);
      __builtin_prefetch(kb + (j0 + 48) * C_, 0, 3);
    }

    // ---- online softmax (row stats per half-wave: xor 8/4/2/1 stays in half)
    float alpha[8];
#pragma unroll
    for (int r = 0; r < 8; ++r) {
      float t = fmaxf(s0[r], s1[r]);
#pragma unroll
      for (int m = 8; m; m >>= 1) t = fmaxf(t, __shfl_xor(t, m, 32));
      const float mn = fmaxf(mrow[r], t);
      alpha[r] = __builtin_amdgcn_exp2f(mrow[r] - mn);
      mrow[r]  = mn;
      const float p0 = __builtin_amdgcn_exp2f(s0[r] - mn);
      const float p1 = __builtin_amdgcn_exp2f(s1[r] - mn);
      pst[r * 32]      = (_Float16)p0;          // LDS transpose bounce
      pst[r * 32 + 16] = (_Float16)p1;
      float rs = p0 + p1;
#pragma unroll
      for (int m = 8; m; m >>= 1) rs += __shfl_xor(rs, m, 32);
      lrow[r] = lrow[r] * alpha[r] + rs;
    }
#pragma unroll
    for (int ct = 0; ct < 16; ++ct)
#pragma unroll
      for (int r = 0; r < 8; ++r) acc[ct][r] *= alpha[r];

    // wave-private LDS region: only need DS ordering, no block barrier
    asm volatile("s_wait_dscnt 0x0" ::: "memory");
    v8h plo = *(const v8h*)pld;
    v8h phi = *(const v8h*)(pld + 16);
    v16h pf;
#pragma unroll
    for (int i = 0; i < 8; ++i) { pf[i] = plo[i]; pf[i + 8] = phi[i]; }

    // ---- PV: P(16x32) x V(32x16) for all 16 channel tiles
#pragma unroll
    for (int ct = 0; ct < 16; ++ct)
      acc[ct] = wmma16(pf, load_b(vb + ct * 16 * HW_ + j0, HW_), acc[ct]);
  }

  float inv[8];
#pragma unroll
  for (int r = 0; r < 8; ++r) inv[r] = 1.f / lrow[r];
  const int mh = (lane >> 4) << 3;
#pragma unroll
  for (int ct = 0; ct < 16; ++ct) {
    _Float16* ob = o2 + (b * HW_ + i0 + mh) * C_ + ct * 16 + (lane & 15);
#pragma unroll
    for (int r = 0; r < 8; ++r)
      ob[r * C_] = (_Float16)(acc[ct][r] * inv[r]);
  }
}

// ---------------------------------------------------------------- kernel 4: output projection + residual

__global__ void __launch_bounds__(256)
__attribute__((amdgpu_waves_per_eu(2)))
k_proj(const _Float16* __restrict__ w16,
       const _Float16* __restrict__ o2,
       const float* __restrict__ bo,
       const float* __restrict__ x,
       float* __restrict__ out) {
  const int b    = blockIdx.x >> 3;
  const int nt   = blockIdx.x & 7;
  const int w    = threadIdx.x >> 5;
  const int lane = threadIdx.x & 31;
  const int n0   = nt * 128 + w * 16;

  const _Float16* ob = o2 + (b * HW_ + n0) * C_;
  const _Float16* wo = w16 + 196608;
  v16h bf[8];
#pragma unroll
  for (int kt = 0; kt < 8; ++kt) bf[kt] = load_b(ob + kt * 32, C_);

  const int nn = n0 + (lane & 15);
  const int mh = (lane >> 4) << 3;
  for (int ot = 0; ot < 16; ++ot) {
    v8f acc = {};
#pragma unroll
    for (int kt = 0; kt < 8; ++kt)
      acc = wmma16(load_a(wo + ot * 16 * C_ + kt * 32, C_), bf[kt], acc);
#pragma unroll
    for (int r = 0; r < 8; ++r) {
      const int o = ot * 16 + mh + r;
      const int idx = (b * C_ + o) * HW_ + nn;
      out[idx] = acc[r] + bo[o] + x[idx];
    }
  }
}

// ---------------------------------------------------------------- launch

extern "C" void kernel_launch(void* const* d_in, const int* in_sizes, int n_in,
                              void* d_out, int out_size, void* d_ws, size_t ws_size,
                              hipStream_t stream) {
  const float* x  = (const float*)d_in[0];
  const float* gs = (const float*)d_in[1];
  const float* gb = (const float*)d_in[2];
  const float* Wq = (const float*)d_in[3];
  const float* bq = (const float*)d_in[4];
  const float* Wk = (const float*)d_in[5];
  const float* bk = (const float*)d_in[6];
  const float* Wv = (const float*)d_in[7];
  const float* bv = (const float*)d_in[8];
  const float* Wo = (const float*)d_in[9];
  const float* bo = (const float*)d_in[10];
  float* out = (float*)d_out;

  // workspace carve-up (halves): weights + 5 activation tensors (~84 MB)
  _Float16* w16 = (_Float16*)d_ws;
  _Float16* hn  = w16 + 262144;
  _Float16* q   = hn  + 8388608;
  _Float16* k   = q   + 8388608;
  _Float16* v   = k   + 8388608;
  _Float16* o2  = v   + 8388608;

  k_w2h      <<<256,  256, 0, stream>>>(Wq, Wk, Wv, Wo, w16);
  k_groupnorm<<<1024, 256, 0, stream>>>(x, gs, gb, hn);
  k_qkv      <<<256,  256, 0, stream>>>(w16, hn, bq, bk, bv, q, k, v);
  k_attn     <<<512,  128, 0, stream>>>(q, k, v, o2);
  k_proj     <<<256,  256, 0, stream>>>(w16, o2, bo, x, out);
}
